// LSTMModuel_66554813218852
// MI455X (gfx1250) — matrix operand order repaired and checked by reading, not compiled, not run on hardware
//
#include <hip/hip_runtime.h>

// ---------------------------------------------------------------------------
// Types
// ---------------------------------------------------------------------------
typedef __attribute__((ext_vector_type(16))) __bf16 v16bf;
typedef __attribute__((ext_vector_type(8)))  float  v8f;
typedef __attribute__((ext_vector_type(4)))  int    v4i;

struct __attribute__((aligned(16))) U4 { unsigned int x, y, z, w; };

union Frag { v16bf v; U4 q[2]; };

#define BM 128
#define BN 128
#define BK 32
#define LDS_STRIDE 40   // elements per LDS row (32 data + 8 pad, 16B-multiple)

static __device__ __forceinline__ float sigmoidf_(float x) {
    return 1.0f / (1.0f + __expf(-x));
}

// ---------------------------------------------------------------------------
// CDNA5 async global->LDS copy (ASYNCcnt-tracked), with safe fallback
// ---------------------------------------------------------------------------
#if __has_builtin(__builtin_amdgcn_global_load_async_to_lds_b128)
#define HAS_ASYNC_LDS 1
#else
#define HAS_ASYNC_LDS 0
#endif

static __device__ __forceinline__ void async_copy16(const void* g, void* l) {
#if HAS_ASYNC_LDS
    __builtin_amdgcn_global_load_async_to_lds_b128(
        (__attribute__((address_space(1))) v4i*)g,
        (__attribute__((address_space(3))) v4i*)l, 0, 0);
#else
    *(U4*)l = *(const U4*)g;
#endif
}

static __device__ __forceinline__ void wait_async0() {
#if HAS_ASYNC_LDS
#if __has_builtin(__builtin_amdgcn_s_wait_asynccnt)
    __builtin_amdgcn_s_wait_asynccnt(0);
#else
    asm volatile("s_wait_asynccnt 0x0" ::: "memory");
#endif
#endif
}

// ---------------------------------------------------------------------------
// Generic bf16 GEMM:  out[M,N] = act(Acat[M,K0+K1] * Bw[N,K]^T + bias) (+resid)
//   A0: [M,K0] bf16 row-major, A1: [M,K1] bf16 row-major (may be null, K1=0)
//   Bw: [N,K0+K1] bf16 row-major; K0,K1 multiples of 32; N multiple of 128
//   Double-buffered LDS staging via GLOBAL_LOAD_ASYNC_TO_LDS_B128.
// ---------------------------------------------------------------------------
template <bool RELU>
__global__ __launch_bounds__(256)
void gemm_bf16(const __bf16* __restrict__ A0, int K0,
               const __bf16* __restrict__ A1, int K1,
               const __bf16* __restrict__ Bw,
               const float* __restrict__ bias,
               const float* __restrict__ resid,
               float* __restrict__ outF,
               __bf16* __restrict__ outH,
               int M, int N)
{
    const int K = K0 + K1;
    __shared__ __bf16 As[2][BM * LDS_STRIDE];
    __shared__ __bf16 Bs[2][BN * LDS_STRIDE];

    const int tid  = threadIdx.x;
    const int lane = tid & 31;
    const int wave = tid >> 5;
    const int wm   = wave & 1;   // 2 waves along M (64 rows each)
    const int wn   = wave >> 1;  // 4 waves along N (32 cols each)

    const int bm = blockIdx.x;
    const int bn = blockIdx.y;

    v8f acc[4][2];
    #pragma unroll
    for (int i = 0; i < 4; ++i)
        #pragma unroll
        for (int j = 0; j < 2; ++j)
            #pragma unroll
            for (int r = 0; r < 8; ++r) acc[i][j][r] = 0.0f;

    // staging coords: 256 threads load 128 rows x 32 cols (each thread 32B)
    const int lrow = tid >> 1;            // 0..127
    const int lseg = (tid & 1) * 16;      // k sub-offset in elements
    const int grow = bm * BM + lrow;      // global A row
    const int gn   = bn * BN + lrow;      // global B row (always < N)
    const bool avalid = (grow < M);

    // per-thread base pointers (computed once)
    const __bf16* aBase0 = A0 + (size_t)grow * K0 + lseg;
    const __bf16* aBase1 = (K1 > 0) ? (A1 + (size_t)grow * K1 + lseg) : nullptr;
    const __bf16* bBase  = Bw + (size_t)gn * K + lseg;
    const int nt0 = K0 / BK;
    const int numT = (K0 + K1) / BK;

    // issue async staging for K-tile kt into LDS buffer buf
    auto issue = [&](int kt, int buf) {
        const __bf16* asrc = (kt < nt0) ? (aBase0 + (size_t)kt * BK)
                                        : (aBase1 + (size_t)(kt - nt0) * BK);
        __bf16* adst = &As[buf][lrow * LDS_STRIDE + lseg];
        if (avalid) {
            async_copy16(asrc,     adst);
            async_copy16(asrc + 8, adst + 8);
        }
        const __bf16* bsrc = bBase + (size_t)kt * BK;
        __bf16* bdst = &Bs[buf][lrow * LDS_STRIDE + lseg];
        async_copy16(bsrc,     bdst);
        async_copy16(bsrc + 8, bdst + 8);
    };

    // fragment coordinates
    const int fr    = lane & 15;
    const int aKoff = (lane >> 4) * 16;  // A: contiguous 16 K per lane
    const int bKoff = (lane >> 4) * 16;  // B: contiguous 16 K per lane

    issue(0, 0);
    wait_async0();
    __syncthreads();

    for (int kt = 0; kt < numT; ++kt) {
        const int buf = kt & 1;
        if (kt + 1 < numT) issue(kt + 1, buf ^ 1);  // prefetch next tile

        Frag a[4], b[2];
        #pragma unroll
        for (int mt = 0; mt < 4; ++mt) {
            const int row = wm * 64 + mt * 16 + fr;
            const U4* p0 = (const U4*)(&As[buf][row * LDS_STRIDE + aKoff]);
            const U4* p1 = (const U4*)(&As[buf][row * LDS_STRIDE + aKoff + 8]);
            a[mt].q[0] = p0[0];
            a[mt].q[1] = p1[0];
        }
        #pragma unroll
        for (int nt = 0; nt < 2; ++nt) {
            const int row = wn * 32 + nt * 16 + fr;
            const U4* p = (const U4*)(&Bs[buf][row * LDS_STRIDE + bKoff]);
            b[nt].q[0] = p[0];
            b[nt].q[1] = p[1];
        }

        #pragma unroll
        for (int mt = 0; mt < 4; ++mt)
            #pragma unroll
            for (int nt = 0; nt < 2; ++nt)
                acc[mt][nt] = __builtin_amdgcn_wmma_f32_16x16x32_bf16(
                    false, a[mt].v, false, b[nt].v,
                    (short)0, acc[mt][nt], false, false);

        wait_async0();     // this wave's prefetch for buf^1 has landed
        __syncthreads();   // whole WG done reading buf / async writes visible
    }

    // ---- epilogue ----
    #pragma unroll
    for (int mt = 0; mt < 4; ++mt) {
        #pragma unroll
        for (int nt = 0; nt < 2; ++nt) {
            const int col = bn * BN + wn * 32 + nt * 16 + (lane & 15);
            const int rowBase = bm * BM + wm * 64 + mt * 16 + ((lane >> 4) << 3);
            const float bv = bias[col];
            #pragma unroll
            for (int r = 0; r < 8; ++r) {
                const int row = rowBase + r;
                if (row >= M) break;
                float v = acc[mt][nt][r] + bv;
                if (RELU) v = fmaxf(v, 0.0f);
                if (resid) v += resid[(size_t)row * N + col];
                if (outF) outF[(size_t)row * N + col] = v;
                if (outH) outH[(size_t)row * N + col] = (__bf16)v;
            }
        }
    }
}

// ---------------------------------------------------------------------------
// Helpers
// ---------------------------------------------------------------------------
__global__ void k_zero16(U4* dst, long n16) {
    long i = (long)blockIdx.x * blockDim.x + threadIdx.x;
    if (i < n16) { U4 z = {0,0,0,0}; dst[i] = z; }
}

// f32 [R,C] -> bf16 [R,CP] (zero padded cols)
__global__ void k_pad_bf16(const float* __restrict__ src, __bf16* __restrict__ dst,
                           int R, int C, int CP) {
    long i = (long)blockIdx.x * blockDim.x + threadIdx.x;
    if (i >= (long)R * CP) return;
    int r = (int)(i / CP), c = (int)(i % CP);
    float v = (c < C) ? src[(size_t)r * C + c] : 0.0f;
    dst[i] = (__bf16)v;
}

// Wc[g, 0:512]=Wih[g], Wc[g, 512:1024]=Whh[g]
__global__ void k_concat_w(const float* __restrict__ Wih,
                           const float* __restrict__ Whh,
                           __bf16* __restrict__ Wc) {
    long i = (long)blockIdx.x * blockDim.x + threadIdx.x;
    if (i >= (long)2048 * 1024) return;
    int g = (int)(i >> 10), k = (int)(i & 1023);
    float v = (k < 512) ? Wih[(size_t)g * 512 + k] : Whh[(size_t)g * 512 + (k - 512)];
    Wc[i] = (__bf16)v;
}

__global__ void k_bias_sum(const float* a, const float* b, float* o, int n) {
    int i = blockIdx.x * blockDim.x + threadIdx.x;
    if (i < n) o[i] = a[i] + b[i];
}

// starts[b] = exclusive prefix sum of sizes (B = 1024, one block)
__global__ void k_seg_starts(const int* __restrict__ sizes, int* __restrict__ starts) {
    __shared__ int tmp[1024];
    int t = threadIdx.x;
    tmp[t] = sizes[t];
    __syncthreads();
    for (int off = 1; off < 1024; off <<= 1) {
        int v = (t >= off) ? tmp[t - off] : 0;
        __syncthreads();
        tmp[t] += v;
        __syncthreads();
    }
    starts[t] = tmp[t] - sizes[t];
}

// scatter ragged bf16 tokens [T,512] -> padded [128,1024,512]
__global__ void k_scatter(const __bf16* __restrict__ enc,
                          const int* __restrict__ starts,
                          const int* __restrict__ sizes,
                          __bf16* __restrict__ Xpad) {
    const int b = blockIdx.x;
    const int start = starts[b];
    const int n = sizes[b];
    const unsigned int* src = (const unsigned int*)(enc + (size_t)start * 512);
    for (int p = 0; p < n; ++p) {
        unsigned int* dst = (unsigned int*)(Xpad + ((size_t)p * 1024 + b) * 512);
        dst[threadIdx.x] = src[(size_t)p * 256 + threadIdx.x];
    }
}

// LSTM cell update: gates [1024,2048] f32, c [1024,512] f32 in/out
__global__ void k_lstm_cell(const float* __restrict__ gates,
                            float* __restrict__ c,
                            __bf16* __restrict__ hbf,
                            float* __restrict__ hfp) {
    int idx = blockIdx.x * blockDim.x + threadIdx.x;   // 0 .. 1024*512-1
    int b = idx >> 9, j = idx & 511;
    const float* g = gates + (size_t)b * 2048;
    float gi = sigmoidf_(g[j]);
    float gf = sigmoidf_(g[j + 512]);
    float gg = tanhf(g[j + 1024]);
    float go = sigmoidf_(g[j + 1536]);
    float cc = gf * c[idx] + gi * gg;
    c[idx] = cc;
    float h = go * tanhf(cc);
    hbf[idx] = (__bf16)h;
    hfp[idx] = h;
}

// out[b] = dot(r2[b,:512], Wd) + bd  (one wave per row)
__global__ __launch_bounds__(256)
void k_decoder(const float* __restrict__ r2, const float* __restrict__ Wd,
               const float* __restrict__ bd, float* __restrict__ out) {
    int wave = blockIdx.x * (blockDim.x >> 5) + (threadIdx.x >> 5);
    int lane = threadIdx.x & 31;
    if (wave >= 1024) return;
    const float* row = r2 + (size_t)wave * 512;
    float s = 0.0f;
    for (int k = lane; k < 512; k += 32) s += row[k] * Wd[k];
    for (int off = 16; off > 0; off >>= 1) s += __shfl_down(s, off, 32);
    if (lane == 0) out[wave] = s + bd[0];
}

// ---------------------------------------------------------------------------
// Host launcher
// ---------------------------------------------------------------------------
static inline int cdiv(long a, long b) { return (int)((a + b - 1) / b); }

extern "C" void kernel_launch(void* const* d_in, const int* in_sizes, int n_in,
                              void* d_out, int out_size, void* d_ws, size_t ws_size,
                              hipStream_t stream) {
    (void)n_in; (void)out_size; (void)ws_size;
    const int*   seg   = (const int*)  d_in[0];
    const float* feats = (const float*)d_in[1];
    const float* We0   = (const float*)d_in[2];
    const float* be0   = (const float*)d_in[3];
    const float* We1   = (const float*)d_in[4];
    const float* be1   = (const float*)d_in[5];
    const float* Wih   = (const float*)d_in[6];
    const float* bih   = (const float*)d_in[7];
    const float* Whh   = (const float*)d_in[8];
    const float* bhh   = (const float*)d_in[9];
    const float* Wl0   = (const float*)d_in[10];
    const float* bl0   = (const float*)d_in[11];
    const float* Wl1   = (const float*)d_in[12];
    const float* bl1   = (const float*)d_in[13];
    const float* Wd    = (const float*)d_in[14];
    const float* bd    = (const float*)d_in[15];
    float* out = (float*)d_out;

    const int T = in_sizes[1] / 164;

    char* ws = (char*)d_ws;
    size_t off = 0;
    auto alloc = [&](size_t bytes) -> void* {
        void* p = ws + off;
        off += (bytes + 255) & ~(size_t)255;
        return p;
    };

    __bf16* featbf = (__bf16*)alloc((size_t)T * 192 * 2);
    __bf16* enc1   = (__bf16*)alloc((size_t)T * 512 * 2);
    __bf16* enc2   = (__bf16*)alloc((size_t)T * 512 * 2);
    __bf16* Xpad   = (__bf16*)alloc((size_t)128 * 1024 * 512 * 2);
    __bf16* W0b    = (__bf16*)alloc((size_t)512 * 192 * 2);
    __bf16* W1b    = (__bf16*)alloc((size_t)512 * 512 * 2);
    __bf16* Wcb    = (__bf16*)alloc((size_t)2048 * 1024 * 2);
    float*  gbias  = (float*) alloc(2048 * 4);
    __bf16* Wl0b   = (__bf16*)alloc((size_t)512 * 512 * 2);
    __bf16* Wl1b   = (__bf16*)alloc((size_t)512 * 512 * 2);
    int*    starts = (int*)   alloc(1024 * 4);
    float*  gates  = (float*) alloc((size_t)1024 * 2048 * 4);
    float*  cbuf   = (float*) alloc((size_t)1024 * 512 * 4);
    __bf16* hbf    = (__bf16*)alloc((size_t)1024 * 512 * 2);
    float*  hfp    = (float*) alloc((size_t)1024 * 512 * 4);
    float*  r1fp   = (float*) alloc((size_t)1024 * 512 * 4);
    __bf16* r1bf   = (__bf16*)alloc((size_t)1024 * 512 * 2);
    float*  r2fp   = (float*) alloc((size_t)1024 * 512 * 4);

    // ---- weight / input conversions ----
    k_pad_bf16<<<cdiv((long)T * 192, 256), 256, 0, stream>>>(feats, featbf, T, 164, 192);
    k_pad_bf16<<<cdiv((long)512 * 192, 256), 256, 0, stream>>>(We0, W0b, 512, 164, 192);
    k_pad_bf16<<<cdiv((long)512 * 512, 256), 256, 0, stream>>>(We1, W1b, 512, 512, 512);
    k_concat_w<<<cdiv((long)2048 * 1024, 256), 256, 0, stream>>>(Wih, Whh, Wcb);
    k_bias_sum<<<cdiv(2048, 256), 256, 0, stream>>>(bih, bhh, gbias, 2048);
    k_pad_bf16<<<cdiv((long)512 * 512, 256), 256, 0, stream>>>(Wl0, Wl0b, 512, 512, 512);
    k_pad_bf16<<<cdiv((long)512 * 512, 256), 256, 0, stream>>>(Wl1, Wl1b, 512, 512, 512);

    // ---- zero padded input, h, c ----
    k_zero16<<<cdiv((long)128 * 1024 * 512 * 2 / 16, 256), 256, 0, stream>>>((U4*)Xpad, (long)128 * 1024 * 512 * 2 / 16);
    k_zero16<<<cdiv((long)1024 * 512 * 4 / 16, 256), 256, 0, stream>>>((U4*)cbuf, (long)1024 * 512 * 4 / 16);
    k_zero16<<<cdiv((long)1024 * 512 * 2 / 16, 256), 256, 0, stream>>>((U4*)hbf, (long)1024 * 512 * 2 / 16);

    // ---- encoder MLP (WMMA) ----
    {
        dim3 g(cdiv(T, BM), 512 / BN);
        gemm_bf16<true><<<g, 256, 0, stream>>>(featbf, 192, nullptr, 0, W0b,
                                               be0, nullptr, nullptr, enc1, T, 512);
        gemm_bf16<true><<<g, 256, 0, stream>>>(enc1, 512, nullptr, 0, W1b,
                                               be1, nullptr, nullptr, enc2, T, 512);
    }

    // ---- pad_sequence scatter ----
    k_seg_starts<<<1, 1024, 0, stream>>>(seg, starts);
    k_scatter<<<1024, 256, 0, stream>>>(enc2, starts, seg, Xpad);

    // ---- LSTM over 128 steps: gates = [x_t | h] @ [Wih|Whh]^T + (bih+bhh) ----
    {
        dim3 g(1024 / BM, 2048 / BN);
        for (int t = 0; t < 128; ++t) {
            const __bf16* xt = Xpad + (size_t)t * 1024 * 512;
            gemm_bf16<false><<<g, 256, 0, stream>>>(xt, 512, hbf, 512, Wcb,
                                                    gbias, nullptr, gates, nullptr,
                                                    1024, 2048);
            k_lstm_cell<<<(1024 * 512) / 256, 256, 0, stream>>>(gates, cbuf, hbf, hfp);
        }
    }

    // ---- residual MLPs ----
    {
        dim3 g(1024 / BM, 512 / BN);
        gemm_bf16<true><<<g, 256, 0, stream>>>(hbf, 512, nullptr, 0, Wl0b,
                                               bl0, hfp, r1fp, r1bf, 1024, 512);
        gemm_bf16<true><<<g, 256, 0, stream>>>(r1bf, 512, nullptr, 0, Wl1b,
                                               bl1, r1fp, r2fp, nullptr, 1024, 512);
    }

    // ---- decoder ----
    k_decoder<<<128, 256, 0, stream>>>(r2fp, Wd, bd, out);
}
